// LSTM_84387517431882
// MI455X (gfx1250) — matrix operand order repaired and checked by reading, not compiled, not run on hardware
//
#include <hip/hip_runtime.h>
#include <math.h>

// MI455X / gfx1250, wave32. Persistent fused 2-layer LSTM using
// v_wmma_f32_16x16x32_f16. ~0.96 TFLOP total, latency-bound recurrence:
// one device-wide step loop with software grid barriers instead of 1024
// kernel launches. Weights (~7MB f16) stay L2-resident (192MB L2).

typedef __attribute__((ext_vector_type(16))) _Float16 v16h;
typedef __attribute__((ext_vector_type(8)))  float    v8f;
typedef __attribute__((ext_vector_type(4)))  unsigned uint4v;

#define B_   256
#define T_   512
#define D_   256
#define H_   512
#define G4   2048        // 4*H gate width
#define NWG  64          // persistent workgroups (must be co-resident)
#define NTHR 256         // 8 waves of 32

// ---------------------------------------------------------------------------
// Prep kernels
// ---------------------------------------------------------------------------

// x [B][T][D] f32  ->  xh [T][B][D] f16  (step-t slice contiguous)
__global__ void cvt_x_kernel(const float* __restrict__ x, _Float16* __restrict__ xh) {
  size_t e = (size_t)blockIdx.x * 256 + threadIdx.x;   // B*T*D = 33,554,432
  int d = (int)(e & (D_ - 1));
  size_t rest = e >> 8;
  int t = (int)(rest & (T_ - 1));
  int b = (int)(rest >> 9);
  xh[((size_t)t * B_ + b) * D_ + d] = (_Float16)x[((size_t)b * T_ + t) * D_ + d];
}

// Pack [Wih0 ; Whh0] as B-matrix: Wpk0[k2][n] = dword{W[n][2k2], W[n][2k2+1]} (f16 pair)
// k in [0,256) -> Wih0 (stride D_), k in [256,768) -> Whh0 (stride H_)
__global__ void pack_w0_kernel(const float* __restrict__ Wih0,
                               const float* __restrict__ Whh0,
                               unsigned* __restrict__ Wpk0) {
  int e = blockIdx.x * 256 + threadIdx.x;   // 384*2048
  int n  = e & (G4 - 1);
  int k2 = e >> 11;                          // 0..383
  int k  = k2 << 1;
  float lo, hi;
  if (k < D_) { lo = Wih0[(size_t)n * D_ + k];        hi = Wih0[(size_t)n * D_ + k + 1]; }
  else        { lo = Whh0[(size_t)n * H_ + (k - D_)]; hi = Whh0[(size_t)n * H_ + (k - D_) + 1]; }
  union { _Float16 h[2]; unsigned u; } pk;
  pk.h[0] = (_Float16)lo; pk.h[1] = (_Float16)hi;
  Wpk0[(size_t)k2 * G4 + n] = pk.u;
}

// Pack [Wih1 ; Whh1]: k in [0,512) -> Wih1, [512,1024) -> Whh1 (both stride H_)
__global__ void pack_w1_kernel(const float* __restrict__ Wih1,
                               const float* __restrict__ Whh1,
                               unsigned* __restrict__ Wpk1) {
  int e = blockIdx.x * 256 + threadIdx.x;   // 512*2048
  int n  = e & (G4 - 1);
  int k2 = e >> 11;                          // 0..511
  int k  = k2 << 1;
  float lo, hi;
  if (k < H_) { lo = Wih1[(size_t)n * H_ + k];        hi = Wih1[(size_t)n * H_ + k + 1]; }
  else        { lo = Whh1[(size_t)n * H_ + (k - H_)]; hi = Whh1[(size_t)n * H_ + (k - H_) + 1]; }
  union { _Float16 h[2]; unsigned u; } pk;
  pk.h[0] = (_Float16)lo; pk.h[1] = (_Float16)hi;
  Wpk1[(size_t)k2 * G4 + n] = pk.u;
}

__global__ void make_bias_kernel(const float* __restrict__ bih0, const float* __restrict__ bhh0,
                                 const float* __restrict__ bih1, const float* __restrict__ bhh1,
                                 float* __restrict__ bias0, float* __restrict__ bias1) {
  int i = blockIdx.x * 256 + threadIdx.x;   // 0..4095
  if (i < G4) bias0[i] = bih0[i] + bhh0[i];
  else        bias1[i - G4] = bih1[i - G4] + bhh1[i - G4];
}

// ---------------------------------------------------------------------------
// Branch-free activations on raw gfx1250 transcendentals (no OCML, no EXEC
// divergence, short critical path after each grid barrier).
// ---------------------------------------------------------------------------
#define LOG2E_ 1.4426950408889634f

__device__ __forceinline__ float fast_sigmoid(float x) {
  // 1/(1+2^(-x*log2e)): saturates correctly at +-inf, branch-free.
  float e = __builtin_amdgcn_exp2f(-x * LOG2E_);
  return __builtin_amdgcn_rcpf(1.0f + e);
}

__device__ __forceinline__ float fast_tanh(float x) {
#if __has_builtin(__builtin_amdgcn_tanhf)
  return __builtin_amdgcn_tanhf(x);        // V_TANH_F32
#elif __has_builtin(__builtin_amdgcn_tanh_f32)
  return __builtin_amdgcn_tanh_f32(x);
#else
  // tanh(x) = copysign((1-e)/(1+e), x), e = 2^(-2|x|*log2e); e in (0,1] so
  // no overflow, underflow -> 0 -> result 1. Branch-free.
  float ax = __builtin_fabsf(x);
  float e  = __builtin_amdgcn_exp2f(ax * (-2.0f * LOG2E_));
  float r  = (1.0f - e) * __builtin_amdgcn_rcpf(1.0f + e);
  return __builtin_copysignf(r, x);
#endif
}

// ---------------------------------------------------------------------------
// Persistent LSTM kernel
// ---------------------------------------------------------------------------

__device__ __forceinline__ v8f wmma16(v16h a, v16h b, v8f c) {
  // (neg_a, A, neg_b, B, c_mod, C, reuse_a, reuse_b)
  return __builtin_amdgcn_wmma_f32_16x16x32_f16(false, a, false, b, (short)0, c, false, false);
}

// A fragment (16x32 f16, M = lane&15): this code's K order — VGPR v, half = lane>>4:
// K = 2v + 16*half, pairs contiguous in row-major A.
// Virtual A row = [p1(row, 0..k1) ; p2(row, 0..)], K-slice uniform per source.
// (Compiler vectorizes the 8 dword loads into 2x global_load_b128.)
__device__ __forceinline__ v16h load_afrag(const _Float16* __restrict__ p1, int s1, int k1,
                                           const _Float16* __restrict__ p2,
                                           int row, int kbase, int half) {
  union { v16h v; unsigned u[8]; } a;
#pragma unroll
  for (int vv = 0; vv < 8; ++vv) {
    int k = kbase + 2 * vv + 16 * half;
    const _Float16* src = (k < k1) ? (p1 + (size_t)row * s1 + k)
                                   : (p2 + (size_t)row * H_ + (k - k1));
    a.u[vv] = *(const unsigned*)src;   // dword = f16 pair (K, K+1)
  }
  return a.v;
}

// Stage one K=32 slice of the 4-gate weight tile into LDS.
// LDS layout Bs[col][kp], col = gate*32 + jj (128 cols), kp = 0..15 pair index,
// row stride 20 dwords -> 16B-aligned b128 fragment reads, conflict-free.
__device__ __forceinline__ void stage_w(const unsigned* __restrict__ Wpk, int k2base,
                                        int nhBase, unsigned* Bs) {
  int tid = threadIdx.x;
#pragma unroll
  for (int i = 0; i < 8; ++i) {
    int idx  = i * 256 + tid;          // 2048 dwords total
    int c    = idx & 127;
    int kp   = idx >> 7;
    int gate = c >> 5;
    int jj   = c & 31;
    int n    = gate * H_ + nhBase + jj;
    Bs[c * 20 + kp] = Wpk[(size_t)(k2base + kp) * G4 + n];
  }
}

// B fragment (32x16 f16): VGPR v holds K pair k2 = v + 8*half.
// Lane's 8 dwords are contiguous in LDS -> two ds_load_b128.
__device__ __forceinline__ v16h load_bfrag(const unsigned* Bs, int c, int half) {
  union { v16h v; uint4v q[2]; } b;
  const uint4v* p = (const uint4v*)(Bs + c * 20 + 8 * half);
  b.q[0] = p[0];
  b.q[1] = p[1];
  return b.v;
}

// One layer step for this wave's 16(M)x16(hidden) tile: 4 gate accumulators,
// K-loop over [p1 ; p2] vs packed weights, then the LSTM cell pointwise update
// done directly on the WMMA C/D register layout (M = r + 8*(lane>>4), N = lane&15).
__device__ __forceinline__ void lstm_phase(
    const _Float16* __restrict__ p1, int s1, int k1,
    const _Float16* __restrict__ p2,
    const unsigned* __restrict__ Wpk, const float* __restrict__ bias,
    float* __restrict__ cbuf, _Float16* __restrict__ hout,
    unsigned* Bs, int rowBase, int nhBase, int colgrp, int Ktot) {
  const int lane = threadIdx.x & 31;
  const int half = lane >> 4;
  const int n15  = lane & 15;

  v8f zero = {0.f, 0.f, 0.f, 0.f, 0.f, 0.f, 0.f, 0.f};
  v8f acc[4];
#pragma unroll
  for (int g = 0; g < 4; ++g) acc[g] = zero;

  const int nslices = Ktot >> 5;
  for (int s = 0; s < nslices; ++s) {
    __syncthreads();                       // previous slice fully consumed
    stage_w(Wpk, s << 4, nhBase, Bs);
    __syncthreads();
    v16h a = load_afrag(p1, s1, k1, p2, rowBase + n15, s << 5, half);
#pragma unroll
    for (int g = 0; g < 4; ++g) {
      v16h b = load_bfrag(Bs, g * 32 + colgrp * 16 + n15, half);
      acc[g] = wmma16(a, b, acc[g]);       // v_wmma_f32_16x16x32_f16
    }
  }

  // Epilogue: i,f,g,o for element (row, col) live at the same (r,lane) slot.
  const int col = nhBase + colgrp * 16 + n15;       // hidden index j
  const float bi = bias[0 * H_ + col];
  const float bf = bias[1 * H_ + col];
  const float bg = bias[2 * H_ + col];
  const float bo = bias[3 * H_ + col];
  const int row0 = rowBase + 8 * half;
#pragma unroll
  for (int r = 0; r < 8; ++r) {
    const int row = row0 + r;                        // batch index
    const size_t off = (size_t)row * H_ + col;
    float iv = fast_sigmoid(acc[0][r] + bi);
    float fv = fast_sigmoid(acc[1][r] + bf);
    float gv = fast_tanh(acc[2][r] + bg);
    float ov = fast_sigmoid(acc[3][r] + bo);
    float cv = fv * cbuf[off] + iv * gv;             // c owned exclusively by this wave
    cbuf[off] = cv;
    hout[off] = (_Float16)(ov * fast_tanh(cv));
  }
}

__device__ __forceinline__ void grid_barrier(unsigned* cnt, unsigned* gen) {
  __syncthreads();
  __threadfence();   // release our h/c stores device-wide
  if (threadIdx.x == 0) {
    unsigned g = __hip_atomic_load(gen, __ATOMIC_RELAXED, __HIP_MEMORY_SCOPE_AGENT);
    unsigned arrived =
        __hip_atomic_fetch_add(cnt, 1u, __ATOMIC_ACQ_REL, __HIP_MEMORY_SCOPE_AGENT);
    if (arrived == NWG - 1) {
      __hip_atomic_store(cnt, 0u, __ATOMIC_RELAXED, __HIP_MEMORY_SCOPE_AGENT);
      __hip_atomic_fetch_add(gen, 1u, __ATOMIC_ACQ_REL, __HIP_MEMORY_SCOPE_AGENT);
    } else {
      while (__hip_atomic_load(gen, __ATOMIC_ACQUIRE, __HIP_MEMORY_SCOPE_AGENT) == g) {
        __builtin_amdgcn_s_sleep(2);
      }
    }
  }
  __syncthreads();
  __threadfence();   // acquire: see other WGPs' stores
}

__global__ __launch_bounds__(NTHR, 1) void lstm_persistent_kernel(
    const _Float16* __restrict__ xh,
    const unsigned* __restrict__ Wpk0, const unsigned* __restrict__ Wpk1,
    const float* __restrict__ bias0, const float* __restrict__ bias1,
    _Float16* h0buf, _Float16* h1buf, float* c0, float* c1,
    unsigned* barrier_state) {
  __shared__ __align__(16) unsigned Bs[128 * 20];   // 10.2 KB of the 320 KB WGP LDS

  const int wave   = threadIdx.x >> 5;
  const int strip  = wave & 3;           // M strip within WG (4 x 16 rows)
  const int colgrp = wave >> 2;          // 0/1: 16-col group within WG's 32 hidden cols
  const int Mtile  = blockIdx.x & 3;     // 4 M tiles of 64 batch rows
  const int NhTile = blockIdx.x >> 2;    // 16 hidden tiles of 32 cols
  const int rowBase = Mtile * 64 + strip * 16;
  const int nhBase  = NhTile * 32;

  unsigned* cnt = barrier_state;
  unsigned* gen = barrier_state + 16;

  for (int t = 0; t < T_; ++t) {
    const int cur = t & 1, prev = cur ^ 1;
    _Float16* h0prev = h0buf + (size_t)prev * B_ * H_;
    _Float16* h0cur  = h0buf + (size_t)cur  * B_ * H_;
    _Float16* h1prev = h1buf + (size_t)prev * B_ * H_;
    _Float16* h1cur  = h1buf + (size_t)cur  * B_ * H_;

    // Layer 0: gates = [x_t ; h0prev] @ [Wih0;Whh0].T   (K = 768)
    lstm_phase(xh + (size_t)t * B_ * D_, D_, D_, h0prev,
               Wpk0, bias0, c0, h0cur, Bs, rowBase, nhBase, colgrp, D_ + H_);
    grid_barrier(cnt, gen);

    // Layer 1: gates = [h0cur ; h1prev] @ [Wih1;Whh1].T (K = 1024)
    lstm_phase(h0cur, H_, H_, h1prev,
               Wpk1, bias1, c1, h1cur, Bs, rowBase, nhBase, colgrp, 2 * H_);
    grid_barrier(cnt, gen);
  }
}

// ---------------------------------------------------------------------------
// Final FC head: out[b] = dot(h2_final[b], Wfc) + bfc
// ---------------------------------------------------------------------------
__global__ void fc_kernel(const _Float16* __restrict__ hfin, const float* __restrict__ Wfc,
                          const float* __restrict__ bfc, float* __restrict__ out) {
  __shared__ float red[128];
  int b = blockIdx.x;
  float s = 0.f;
  for (int j = threadIdx.x; j < H_; j += 128)
    s += (float)hfin[(size_t)b * H_ + j] * Wfc[j];
  red[threadIdx.x] = s;
  __syncthreads();
  for (int off = 64; off > 0; off >>= 1) {
    if ((int)threadIdx.x < off) red[threadIdx.x] += red[threadIdx.x + off];
    __syncthreads();
  }
  if (threadIdx.x == 0) out[b] = red[0] + bfc[0];
}

// ---------------------------------------------------------------------------
extern "C" void kernel_launch(void* const* d_in, const int* in_sizes, int n_in,
                              void* d_out, int out_size, void* d_ws, size_t ws_size,
                              hipStream_t stream) {
  (void)in_sizes; (void)n_in; (void)out_size; (void)ws_size;

  const float* x    = (const float*)d_in[0];
  const float* Wih0 = (const float*)d_in[1];
  const float* Whh0 = (const float*)d_in[2];
  const float* bih0 = (const float*)d_in[3];
  const float* bhh0 = (const float*)d_in[4];
  const float* Wih1 = (const float*)d_in[5];
  const float* Whh1 = (const float*)d_in[6];
  const float* bih1 = (const float*)d_in[7];
  const float* bhh1 = (const float*)d_in[8];
  const float* Wfc  = (const float*)d_in[9];
  const float* bfc  = (const float*)d_in[10];
  float* out = (float*)d_out;

  char* ws = (char*)d_ws;
  size_t off = 0;
  // --- zero-initialized region (h/c state + barrier), re-zeroed every call ---
  _Float16* h0buf = (_Float16*)(ws + off); off += 2 * (size_t)B_ * H_ * sizeof(_Float16);
  _Float16* h1buf = (_Float16*)(ws + off); off += 2 * (size_t)B_ * H_ * sizeof(_Float16);
  float*    c0    = (float*)(ws + off);    off += (size_t)B_ * H_ * sizeof(float);
  float*    c1    = (float*)(ws + off);    off += (size_t)B_ * H_ * sizeof(float);
  unsigned* bar   = (unsigned*)(ws + off); off += 256;
  size_t zero_bytes = off;
  // --- recomputed-every-call region ---
  _Float16* xh    = (_Float16*)(ws + off); off += (size_t)B_ * T_ * D_ * sizeof(_Float16);
  unsigned* Wpk0  = (unsigned*)(ws + off); off += (size_t)384 * G4 * sizeof(unsigned);
  unsigned* Wpk1  = (unsigned*)(ws + off); off += (size_t)512 * G4 * sizeof(unsigned);
  float*    bias0 = (float*)(ws + off);    off += G4 * sizeof(float);
  float*    bias1 = (float*)(ws + off);    off += G4 * sizeof(float);
  // total ~76.6 MB workspace

  hipMemsetAsync(d_ws, 0, zero_bytes, stream);

  cvt_x_kernel   <<<(B_ * T_ * D_) / 256, 256, 0, stream>>>(x, xh);
  pack_w0_kernel <<<(384 * G4) / 256,     256, 0, stream>>>(Wih0, Whh0, Wpk0);
  pack_w1_kernel <<<(512 * G4) / 256,     256, 0, stream>>>(Wih1, Whh1, Wpk1);
  make_bias_kernel<<<16, 256, 0, stream>>>(bih0, bhh0, bih1, bhh1, bias0, bias1);

  lstm_persistent_kernel<<<NWG, NTHR, 0, stream>>>(xh, Wpk0, Wpk1, bias0, bias1,
                                                   h0buf, h1buf, c0, c1, bar);

  // final layer-1 hidden state lives in parity buffer (T-1)&1 == 1
  fc_kernel<<<B_, 128, 0, stream>>>(h1buf + (size_t)1 * B_ * H_, Wfc, bfc, out);
}
